// AxialAttention_45749991637536
// MI455X (gfx1250) — matrix operand order repaired and checked by reading, not compiled, not run on hardware
//
#include <hip/hip_runtime.h>
#include <hip/hip_bf16.h>

// ---------------------------------------------------------------------------
// Axial attention, fused CDNA5 pipeline.
//  - all GEMMs via v_wmma_f32_16x16x32_f16 (f32 accum)
//  - weight panels staged global->LDS with the Tensor Data Mover
//    (tensor_load_to_lds + s_wait_tensorcnt)
// ---------------------------------------------------------------------------

typedef __attribute__((ext_vector_type(16))) _Float16 v16h;
typedef __attribute__((ext_vector_type(8)))  float    v8f;
typedef __attribute__((ext_vector_type(4)))  unsigned int v4u;
typedef __attribute__((ext_vector_type(8)))  int      v8i_;
typedef __attribute__((ext_vector_type(4)))  int      v4i_;

struct alignas(16) B16 { unsigned long long lo, hi; };   // 16-byte move

// ---- problem constants ----
#define NN_  2
#define CC_  128
#define DD_  8
#define HH_  56
#define WW_  56
#define GG_  8
#define BB_  896          // N*D*W
#define DHW_ 25088        // D*H*W
#define CDHW_ 3211264     // C*D*H*W

// ---- workspace layout (bytes) ----
#define OFF_BN1_SUM   0
#define OFF_BN1_SQ    1024
#define OFF_BN1_SCALE 2048
#define OFF_BN1_SHIFT 3072
#define OFF_BN2_SUM   4096
#define OFF_BN2_SQ    4224
#define OFF_BN2_SCALE 4352
#define OFF_BN2_SHIFT 4480
#define OFF_WQKV      8192ULL          // 256x128 f16
#define OFF_WFC       73728ULL         // 128x256 f16
#define OFF_WM1       139264ULL        // 256x128 f16
#define OFF_WM2       204800ULL        // 128x256 f16
#define OFF_EMB       270336ULL        // 32x56x56 f32
#define OFF_QKV       671744ULL        // 896x256x56 f32
#define OFF_SO2       52051968ULL      // 896x128x56 f32

// ---------------------------------------------------------------------------
// helpers
// ---------------------------------------------------------------------------
__device__ __forceinline__ float wred_sum(float v) {
  for (int o = 16; o > 0; o >>= 1) v += __shfl_xor(v, o, 32);
  return v;
}
__device__ __forceinline__ float wred_max(float v) {
  for (int o = 16; o > 0; o >>= 1) v = fmaxf(v, __shfl_xor(v, o, 32));
  return v;
}
// reduce within each 16-lane half of the wave (same output channel per half)
__device__ __forceinline__ float hred_sum(float v) {
  for (int o = 1; o < 16; o <<= 1) v += __shfl_xor(v, o, 32);
  return v;
}

// TDM: async-copy `ndwords` dwords from global to LDS byte offset `lds_byte_off`.
// D# per CDNA5 ISA: count=1, type=2 (image),
// data_size=2 (4B), 2D tile (ndwords x 1).  Kernels here use only dynamic LDS,
// so the allocation base is 0 and the static byte offset IS the LDS address.
// Toolchain is clang-23 / therock: 6-arg builtin
//   (uint32x4 g0, int32x8 g1, int32x4 g2, int32x4 g3, int32x8 pad, i32 cpol)
__device__ __forceinline__ void tdm_load_1d(const void* gsrc, unsigned lds_byte_off,
                                            unsigned ndwords) {
  unsigned long long ga = (unsigned long long)gsrc;
  v4u g0;
  g0.x = 1u;                                               // count=1, user mode
  g0.y = lds_byte_off;                                     // lds_addr
  g0.z = (unsigned)(ga & 0xffffffffu);                     // global_addr lo
  g0.w = (unsigned)((ga >> 32) & 0x01ffffffu) | (2u << 30);// addr hi | type=2
  v8i_ g1;
  g1[0] = 0x20000;                       // workgroup_mask=0, data_size=2 (4B)
  g1[1] = (int)(ndwords << 16);          // tensor_dim0[15:0] @ bits63:48
  g1[2] = (int)((ndwords >> 16) & 0xffffu) | (1 << 16); // dim0 hi | tensor_dim1=1
  g1[3] = (int)(ndwords << 16);          // tensor_dim1 hi(0) | tile_dim0
  g1[4] = 1;                             // tile_dim1=1, tile_dim2=0
  g1[5] = (int)ndwords;                  // tensor_dim0_stride lo
  g1[6] = 0;
  g1[7] = 0;
  v4i_ z4 = {0, 0, 0, 0};                // groups 2/3 unused (<=2D tensor)
  v8i_ z8 = {0, 0, 0, 0, 0, 0, 0, 0};
  __builtin_amdgcn_tensor_load_to_lds(g0, g1, z4, z4, z8, 0);
}
__device__ __forceinline__ void tdm_wait() {
  __builtin_amdgcn_s_wait_tensorcnt(0);
}

// One 16x16 output tile, K = kchunks*32, f16 inputs from LDS, f32 accum.
// A row-major [M][lda]; B stored transposed [N][ldb] (Bt[n][k]).
// ISA layouts: A lane m=l&15, comps 0..7 -> K=hi*8+j, comps 8..15 -> K=16+hi*8+j
//              B lane n=l&15, comps 0..7 -> K=hi*8+j, comps 8..15 -> K=16+hi*8+j
__device__ __forceinline__ v8f wmma_tile(const _Float16* __restrict__ As, int lda,
                                         const _Float16* __restrict__ Bs, int ldb,
                                         int mt, int nt, int kchunks, int lane) {
  v8f acc = {0.f, 0.f, 0.f, 0.f, 0.f, 0.f, 0.f, 0.f};
  const int l15 = lane & 15;
  const int hi  = lane >> 4;
  const _Float16* arow = As + (mt * 16 + l15) * lda + hi * 8;
  const _Float16* brow = Bs + (nt * 16 + l15) * ldb + hi * 8;
  for (int kc = 0; kc < kchunks; ++kc) {
    v16h a, b;
    B16* ap = reinterpret_cast<B16*>(&a);
    B16* bp = reinterpret_cast<B16*>(&b);
    const _Float16* pa = arow + kc * 32;
    ap[0] = *reinterpret_cast<const B16*>(pa);
    ap[1] = *reinterpret_cast<const B16*>(pa + 16);
    const _Float16* pb = brow + kc * 32;
    bp[0] = *reinterpret_cast<const B16*>(pb);
    bp[1] = *reinterpret_cast<const B16*>(pb + 16);
    acc = __builtin_amdgcn_wmma_f32_16x16x32_f16(false, a, false, b,
                                                 (short)0, acc, false, false);
  }
  return acc;
}

__device__ __forceinline__ void decode_b(int b, int& nn, int& dd, int& wI) {
  nn = b / (DD_ * WW_);
  int rem = b % (DD_ * WW_);
  dd = rem / WW_;
  wI = rem % WW_;
}

// ---------------------------------------------------------------------------
// k0: weights -> f16, build all_emb, zero BN accumulators
// ---------------------------------------------------------------------------
__global__ void k0_prep(const float* __restrict__ wqkv, const float* __restrict__ wfc,
                        const float* __restrict__ wm1, const float* __restrict__ wm2,
                        const float* __restrict__ rel, char* __restrict__ ws) {
  int idx = blockIdx.x * blockDim.x + threadIdx.x;
  int stride = gridDim.x * blockDim.x;
  _Float16* hq = (_Float16*)(ws + OFF_WQKV);
  _Float16* hf = (_Float16*)(ws + OFF_WFC);
  _Float16* h1 = (_Float16*)(ws + OFF_WM1);
  _Float16* h2 = (_Float16*)(ws + OFF_WM2);
  for (int i = idx; i < 256 * 128; i += stride) {
    hq[i] = (_Float16)wqkv[i];
    hf[i] = (_Float16)wfc[i];
    h1[i] = (_Float16)wm1[i];
    h2[i] = (_Float16)wm2[i];
  }
  float* emb = (float*)(ws + OFF_EMB);
  for (int i = idx; i < 32 * 3136; i += stride) {
    int c = i / 3136, r = i % 3136;
    int ii = r / 56, jj = r % 56;
    emb[i] = rel[c * 111 + (ii - jj + 55)];
  }
  float* st = (float*)ws;                       // zero all BN accumulators
  for (int i = idx; i < 1152; i += stride) st[i] = 0.f;
}

// ---------------------------------------------------------------------------
// k1: per-b LayerNorm over C, WMMA qkv GEMM (256x56, K=128), BN1 stats
// ---------------------------------------------------------------------------
__global__ void __launch_bounds__(256)
k1_ln_qkv(const float* __restrict__ x, const float* __restrict__ ln_g,
          const float* __restrict__ ln_b, char* __restrict__ ws) {
  extern __shared__ char smem[];
  _Float16* xh = (_Float16*)smem;                  // Bt: [64][128] (n=h, k=c)
  _Float16* wq = (_Float16*)(smem + 16384);        // A:  [256][128]
  const int b = blockIdx.x;
  const int lane = threadIdx.x & 31, wv = threadIdx.x >> 5;
  int nn, dd, wI; decode_b(b, nn, dd, wI);
  const size_t xbase = (size_t)nn * CDHW_ + (size_t)dd * HH_ * WW_ + wI;

  // kick TDM weight DMA (wave 0), overlap with LayerNorm
  if (wv == 0) tdm_load_1d(ws + OFF_WQKV, 16384u, 16384u);

  // LayerNorm over C for each h: 8 waves x 7 rows
  for (int t = 0; t < 7; ++t) {
    int h = wv * 7 + t;
    float xv[4], s = 0.f;
    for (int j = 0; j < 4; ++j) {
      int c = j * 32 + lane;
      xv[j] = x[xbase + (size_t)c * DHW_ + (size_t)h * WW_];
      s += xv[j];
    }
    s = wred_sum(s);
    float mu = s * (1.f / 128.f);
    float vs = 0.f;
    for (int j = 0; j < 4; ++j) { float d = xv[j] - mu; vs += d * d; }
    vs = wred_sum(vs);
    float inv = rsqrtf(vs * (1.f / 128.f) + 1e-6f);
    for (int j = 0; j < 4; ++j) {
      int c = j * 32 + lane;
      xh[h * 128 + c] = (_Float16)((xv[j] - mu) * inv * ln_g[c] + ln_b[c]);
    }
  }
  if (wv == 0) tdm_wait();
  __syncthreads();

  float* qkv_out = (float*)(ws + OFF_QKV) + (size_t)b * 256 * 56;
  float* s_sum = (float*)(ws + OFF_BN1_SUM);
  float* s_sq  = (float*)(ws + OFF_BN1_SQ);
  // M=256 (16 tiles) x N=64 (4 tiles), K=128 (4 chunks) -> 64 tiles / 8 waves
  for (int t = wv; t < 64; t += 8) {
    int mt = t >> 2, nt = t & 3;
    v8f acc = wmma_tile(wq, 128, xh, 128, mt, nt, 4, lane);
    int n  = nt * 16 + (lane & 15);
    int hi = lane >> 4;
    bool valid = (n < 56);
    for (int r = 0; r < 8; ++r) {
      int o = mt * 16 + r + 8 * hi;
      float v = valid ? acc[r] : 0.f;
      if (valid) qkv_out[o * 56 + n] = v;
      float vv = hred_sum(v);
      float vq = hred_sum(v * v);
      if ((lane & 15) == 0) { atomicAdd(&s_sum[o], vv); atomicAdd(&s_sq[o], vq); }
    }
  }
}

// ---------------------------------------------------------------------------
// k2 / k4: BN finalize
// ---------------------------------------------------------------------------
__global__ void k2_bn1(const float* __restrict__ g, const float* __restrict__ bb,
                       char* __restrict__ ws) {
  int o = threadIdx.x;  // 256
  const float cnt = 896.f * 56.f;
  float* sum = (float*)(ws + OFF_BN1_SUM);
  float* sq  = (float*)(ws + OFF_BN1_SQ);
  float* sc  = (float*)(ws + OFF_BN1_SCALE);
  float* sh  = (float*)(ws + OFF_BN1_SHIFT);
  float mu = sum[o] / cnt;
  float var = sq[o] / cnt - mu * mu;
  float s = g[o] * rsqrtf(var + 1e-5f);
  sc[o] = s;
  sh[o] = bb[o] - mu * s;
}

__global__ void k4_bn2(const float* __restrict__ g, const float* __restrict__ bb,
                       char* __restrict__ ws) {
  int ch = threadIdx.x;
  if (ch >= 24) return;
  const float cnt = 896.f * 3136.f;
  float* sum = (float*)(ws + OFF_BN2_SUM);
  float* sq  = (float*)(ws + OFF_BN2_SQ);
  float* sc  = (float*)(ws + OFF_BN2_SCALE);
  float* sh  = (float*)(ws + OFF_BN2_SHIFT);
  float mu = sum[ch] / cnt;
  float var = sq[ch] / cnt - mu * mu;
  float s = g[ch] * rsqrtf(var + 1e-5f);
  sc[ch] = s;
  sh[ch] = bb[ch] - mu * s;
}

// ---------------------------------------------------------------------------
// k3: per-b qk/qr/kr stats for the similarity BatchNorm (recompute, no store)
// ---------------------------------------------------------------------------
__global__ void __launch_bounds__(256)
k3_sim_stats(char* __restrict__ ws) {
  extern __shared__ char smem[];
  float* qn = (float*)smem;  // [256*56] normalized qkv
  const int b = blockIdx.x;
  const int lane = threadIdx.x & 31;
  {
    const float* qkv = (const float*)(ws + OFF_QKV) + (size_t)b * 256 * 56;
    const float* sc = (const float*)(ws + OFF_BN1_SCALE);
    const float* sh = (const float*)(ws + OFF_BN1_SHIFT);
    for (int i = threadIdx.x; i < 256 * 56; i += 256) {
      int o = i / 56;
      qn[i] = qkv[i] * sc[o] + sh[o];
    }
  }
  __syncthreads();
  const float* emb = (const float*)(ws + OFF_EMB);
  float* bsum = (float*)(ws + OFF_BN2_SUM);
  float* bsq  = (float*)(ws + OFF_BN2_SQ);
  for (int g = 0; g < 8; ++g) {
    const float* qg = qn + g * 32 * 56;
    float aqk = 0, aqr = 0, akr = 0, sqk = 0, sqr = 0, skr = 0;
    for (int idx = threadIdx.x; idx < 3136; idx += 256) {
      int i = idx / 56, j = idx % 56;
      float vqk = 0, vqr = 0, vkr = 0;
      for (int c = 0; c < 8; ++c) {
        float qv = qg[c * 56 + i];
        vqk += qv * qg[(8 + c) * 56 + j];
        vqr += qv * emb[(size_t)c * 3136 + idx];
        vkr += qg[(8 + c) * 56 + i] * emb[(size_t)(8 + c) * 3136 + idx];
      }
      aqk += vqk; sqk += vqk * vqk;
      aqr += vqr; sqr += vqr * vqr;
      akr += vkr; skr += vkr * vkr;
    }
    aqk = wred_sum(aqk); sqk = wred_sum(sqk);
    aqr = wred_sum(aqr); sqr = wred_sum(sqr);
    akr = wred_sum(akr); skr = wred_sum(skr);
    if (lane == 0) {
      atomicAdd(&bsum[g], aqk);      atomicAdd(&bsq[g], sqk);
      atomicAdd(&bsum[8 + g], aqr);  atomicAdd(&bsq[8 + g], sqr);
      atomicAdd(&bsum[16 + g], akr); atomicAdd(&bsq[16 + g], skr);
    }
  }
}

// ---------------------------------------------------------------------------
// k5: per-b attention core + softmax + WMMA sv + sve + WMMA fc GEMM
//     + reshape trick + residual
// LDS map: qn @0 (57344) | sA @57344 (12544) | sB @69888 (12544)
//          sop @82432 (32768) | simh @115200 (8192) | vth @123392 (2048)
//          phase B: wfc f16 @0 (65536, aliases dead qn/sA)
// ---------------------------------------------------------------------------
__global__ void __launch_bounds__(256)
k5_attn_fc(const float* __restrict__ x, char* __restrict__ ws) {
  extern __shared__ char smem[];
  float* qn = (float*)smem;
  float* sA = (float*)(smem + 57344);
  float* sB = (float*)(smem + 69888);
  _Float16* sop  = (_Float16*)(smem + 82432);   // [64][256]
  _Float16* simh = (_Float16*)(smem + 115200);  // [64][64]
  _Float16* vth  = (_Float16*)(smem + 123392);  // [16][64]
  const int b = blockIdx.x;
  const int lane = threadIdx.x & 31, wv = threadIdx.x >> 5;
  {
    const float* qkv = (const float*)(ws + OFF_QKV) + (size_t)b * 256 * 56;
    const float* sc = (const float*)(ws + OFF_BN1_SCALE);
    const float* sh = (const float*)(ws + OFF_BN1_SHIFT);
    for (int i = threadIdx.x; i < 256 * 56; i += 256) {
      int o = i / 56;
      qn[i] = qkv[i] * sc[o] + sh[o];
    }
  }
  const float* emb = (const float*)(ws + OFF_EMB);
  const float* b2sc = (const float*)(ws + OFF_BN2_SCALE);
  const float* b2sh = (const float*)(ws + OFF_BN2_SHIFT);
  __syncthreads();

  for (int g = 0; g < 8; ++g) {
    const float* qg = qn + g * 32 * 56;
    float cqk = b2sc[g], cqr = b2sc[8 + g], ckr = b2sc[16 + g];
    float shift3 = b2sh[g] + b2sh[8 + g] + b2sh[16 + g];
    for (int idx = threadIdx.x; idx < 3136; idx += 256) {
      int i = idx / 56, j = idx % 56;
      float vqk = 0, vqr = 0, vkr = 0;
      for (int c = 0; c < 8; ++c) {
        float qv = qg[c * 56 + i];
        vqk += qv * qg[(8 + c) * 56 + j];
        vqr += qv * emb[(size_t)c * 3136 + idx];
        vkr += qg[(8 + c) * 56 + i] * emb[(size_t)(8 + c) * 3136 + idx];
      }
      sA[idx] = vqk * cqk + vqr * cqr + shift3;  // qk+qr (normalized) + shifts
      sB[idx] = vkr * ckr;                        // kr0 (normalized, pre-transpose)
    }
    __syncthreads();
    // softmax over j of s[i][j] = sA[i][j] + sB[j][i]   (kr transpose folded in)
    for (int t = 0; t < 7; ++t) {
      int i = wv * 7 + t;
      float s0 = sA[i * 56 + lane] + sB[lane * 56 + i];
      float s1 = (lane < 24) ? (sA[i * 56 + 32 + lane] + sB[(32 + lane) * 56 + i])
                             : -1e30f;
      float m = wred_max(fmaxf(s0, s1));
      float e0 = expf(s0 - m);
      float e1 = (lane < 24) ? expf(s1 - m) : 0.f;
      float ssum = wred_sum(e0 + e1);
      float inv = 1.f / ssum;
      sA[i * 56 + lane] = e0 * inv;
      if (lane < 24) sA[i * 56 + 32 + lane] = e1 * inv;
    }
    __syncthreads();
    // pack sim (A, zero-padded K cols) and V^T (Bt, zero-padded K cols) as f16
    for (int idx = threadIdx.x; idx < 64 * 64; idx += 256) {
      int i = idx >> 6, j = idx & 63;
      simh[idx] = (_Float16)((i < 56 && j < 56) ? sA[i * 56 + j] : 0.f);
    }
    for (int idx = threadIdx.x; idx < 16 * 64; idx += 256) {
      int c = idx >> 6, j = idx & 63;
      vth[idx] = (_Float16)((j < 56) ? qg[(16 + c) * 56 + j] : 0.f);
    }
    // sve (embedding factor depends on output row -> not a GEMM): odd channels
    for (int idx = threadIdx.x; idx < 16 * 56; idx += 256) {
      int c = idx / 56, i = idx % 56;
      const float* simrow = &sA[i * 56];
      const float* embc = emb + (size_t)(16 + c) * 3136 + i * 56;
      float asve = 0.f;
      for (int j = 0; j < 56; ++j) asve += simrow[j] * embc[j];
      sop[i * 256 + g * 32 + 2 * c + 1] = (_Float16)asve;
    }
    __syncthreads();
    // sv via WMMA: M=64(i) x N=16(c), K=64 -> even channels
    for (int t = wv; t < 4; t += 8) {
      v8f acc = wmma_tile(simh, 64, vth, 64, t, 0, 2, lane);
      int c = lane & 15, hi = lane >> 4;
      for (int r = 0; r < 8; ++r) {
        int i = t * 16 + r + 8 * hi;
        if (i < 56) sop[i * 256 + g * 32 + 2 * c] = (_Float16)acc[r];
      }
    }
    __syncthreads();
  }

  // phase B: fc GEMM (M=64 h, N=128 o, K=256), residual + reshape trick fused
  const _Float16* wfc = (const _Float16*)smem;   // alias dead qn/sA region
  if (wv == 0) { tdm_load_1d(ws + OFF_WFC, 0u, 16384u); tdm_wait(); }
  __syncthreads();
  int nn, dd, wI; decode_b(b, nn, dd, wI);
  const size_t xbase = (size_t)nn * CDHW_ + (size_t)dd * HH_ * WW_ + wI;
  float* so2 = (float*)(ws + OFF_SO2) + (size_t)b * 128 * 56;
  for (int t = wv; t < 32; t += 8) {
    int mt = t >> 3, nt = t & 7;
    v8f acc = wmma_tile(sop, 256, wfc, 256, mt, nt, 8, lane);
    int n  = nt * 16 + (lane & 15);
    int hi = lane >> 4;
    for (int r = 0; r < 8; ++r) {
      int m = mt * 16 + r + 8 * hi;     // h index of fc output (b,h,o)
      if (m < 56) {
        int flat = m * 128 + n;          // (H,C) flat -> reinterpret as (C,H)
        int c2 = flat / 56, h2 = flat % 56;
        float res = x[xbase + (size_t)c2 * DHW_ + (size_t)h2 * WW_];
        so2[c2 * 56 + h2] = res + acc[r];
      }
    }
  }
}

// ---------------------------------------------------------------------------
// k6: per-b LN + WMMA mlp1(+relu) + WMMA mlp2 + residual + output transpose
// LDS map: so2 @0 (28672) | lnh @28672 (16384) | y1h @45056 (32768)
//          wb @77824 (65536, TDM target, reused for w_mlp1 then w_mlp2)
// ---------------------------------------------------------------------------
__global__ void __launch_bounds__(256)
k6_mlp(const float* __restrict__ ln_g, const float* __restrict__ ln_b,
       char* __restrict__ ws, float* __restrict__ out) {
  extern __shared__ char smem[];
  float*    so2 = (float*)smem;
  _Float16* lnh = (_Float16*)(smem + 28672);
  _Float16* y1h = (_Float16*)(smem + 45056);
  _Float16* wb  = (_Float16*)(smem + 77824);
  const int b = blockIdx.x;
  const int lane = threadIdx.x & 31, wv = threadIdx.x >> 5;

  if (wv == 0) tdm_load_1d(ws + OFF_WM1, 77824u, 16384u);  // overlap with LN

  {
    const float* src = (const float*)(ws + OFF_SO2) + (size_t)b * 128 * 56;
    for (int i = threadIdx.x; i < 128 * 56; i += 256) so2[i] = src[i];
  }
  __syncthreads();
  // LayerNorm over C per h
  for (int t = 0; t < 7; ++t) {
    int h = wv * 7 + t;
    float xv[4], s = 0.f;
    for (int j = 0; j < 4; ++j) { xv[j] = so2[(j * 32 + lane) * 56 + h]; s += xv[j]; }
    s = wred_sum(s);
    float mu = s * (1.f / 128.f);
    float vs = 0.f;
    for (int j = 0; j < 4; ++j) { float d = xv[j] - mu; vs += d * d; }
    vs = wred_sum(vs);
    float inv = rsqrtf(vs * (1.f / 128.f) + 1e-6f);
    for (int j = 0; j < 4; ++j) {
      int c = j * 32 + lane;
      lnh[h * 128 + c] = (_Float16)((xv[j] - mu) * inv * ln_g[c] + ln_b[c]);
    }
  }
  if (wv == 0) tdm_wait();
  __syncthreads();
  // mlp1: M=64(h) x N=256(o), K=128 -> relu -> y1h f16
  for (int t = wv; t < 64; t += 8) {
    int mt = t >> 4, nt = t & 15;
    v8f acc = wmma_tile(lnh, 128, wb, 128, mt, nt, 4, lane);
    int n = nt * 16 + (lane & 15), hi = lane >> 4;
    for (int r = 0; r < 8; ++r) {
      int m = mt * 16 + r + 8 * hi;
      float v = acc[r];
      y1h[m * 256 + n] = (_Float16)(v > 0.f ? v : 0.f);
    }
  }
  __syncthreads();
  if (wv == 0) { tdm_load_1d(ws + OFF_WM2, 77824u, 16384u); tdm_wait(); }
  __syncthreads();
  // mlp2: M=64(h) x N=128(c), K=256; fused residual + NCDHW scatter
  int nn, dd, wI; decode_b(b, nn, dd, wI);
  const size_t obase = (size_t)nn * CDHW_ + (size_t)dd * HH_ * WW_ + wI;
  for (int t = wv; t < 32; t += 8) {
    int mt = t >> 3, nt = t & 7;
    v8f acc = wmma_tile(y1h, 256, wb, 256, mt, nt, 8, lane);
    int n = nt * 16 + (lane & 15), hi = lane >> 4;
    for (int r = 0; r < 8; ++r) {
      int m = mt * 16 + r + 8 * hi;   // h
      if (m < 56) {
        out[obase + (size_t)n * DHW_ + (size_t)m * WW_] = acc[r] + so2[n * 56 + m];
      }
    }
  }
}

// ---------------------------------------------------------------------------
// launch
// ---------------------------------------------------------------------------
extern "C" void kernel_launch(void* const* d_in, const int* in_sizes, int n_in,
                              void* d_out, int out_size, void* d_ws, size_t ws_size,
                              hipStream_t stream) {
  const float* x        = (const float*)d_in[0];
  const float* w_qkv    = (const float*)d_in[1];
  const float* bn_qkv_g = (const float*)d_in[2];
  const float* bn_qkv_b = (const float*)d_in[3];
  const float* ln_g     = (const float*)d_in[4];
  const float* ln_b     = (const float*)d_in[5];
  const float* bn_sim_g = (const float*)d_in[6];
  const float* bn_sim_b = (const float*)d_in[7];
  const float* relative = (const float*)d_in[8];
  const float* w_fc     = (const float*)d_in[9];
  const float* w_mlp1   = (const float*)d_in[10];
  const float* w_mlp2   = (const float*)d_in[11];
  char* ws = (char*)d_ws;
  float* out = (float*)d_out;

  // allow >64KB dynamic LDS (CDNA5 WGP has 320KB)
  (void)hipFuncSetAttribute((const void*)k1_ln_qkv,
      hipFuncAttributeMaxDynamicSharedMemorySize, 81920);
  (void)hipFuncSetAttribute((const void*)k3_sim_stats,
      hipFuncAttributeMaxDynamicSharedMemorySize, 57344);
  (void)hipFuncSetAttribute((const void*)k5_attn_fc,
      hipFuncAttributeMaxDynamicSharedMemorySize, 125440);
  (void)hipFuncSetAttribute((const void*)k6_mlp,
      hipFuncAttributeMaxDynamicSharedMemorySize, 143360);

  k0_prep<<<256, 256, 0, stream>>>(w_qkv, w_fc, w_mlp1, w_mlp2, relative, ws);
  k1_ln_qkv<<<BB_, 256, 81920, stream>>>(x, ln_g, ln_b, ws);
  k2_bn1<<<1, 256, 0, stream>>>(bn_qkv_g, bn_qkv_b, ws);
  k3_sim_stats<<<BB_, 256, 57344, stream>>>(ws);
  k4_bn2<<<1, 32, 0, stream>>>(bn_sim_g, bn_sim_b, ws);
  k5_attn_fc<<<BB_, 256, 125440, stream>>>(x, ws);
  k6_mlp<<<BB_, 256, 143360, stream>>>(ln_g, ln_b, ws, out);
}
